// CQRNLayer_2937757630833
// MI455X (gfx1250) — matrix operand order repaired and checked by reading, not compiled, not run on hardware
//
#include <hip/hip_runtime.h>

// Problem dimensions (fixed by the reference).
#define S_   256
#define B_   16
#define C_   64
#define N_   128
#define H_   64
#define H3_  192
#define TAPS 9
#define NROWS 136   // padded conv columns: 4 | 128 | 4 (stored as LDS rows)
#define TSTR  80    // transposed-LDS row stride in bf16 (rows stay 32B-aligned)

typedef __attribute__((ext_vector_type(16))) __bf16 v16bf;
typedef __attribute__((ext_vector_type(8)))  float  v8f;

// b128 async-copy payload type per the builtin's signature:
// (AS(1) int4* gsrc, AS(3) int4* ldsdst, imm offset, imm cpol)
typedef int v4i_ __attribute__((vector_size(16)));
typedef __attribute__((address_space(1))) v4i_* gptr128_t;
typedef __attribute__((address_space(3))) v4i_* lptr128_t;

// Explicit LDS pointer types so the noinline callee keeps ds_* addressing.
typedef __attribute__((address_space(3))) const __bf16* lds_bf16_cptr;
typedef __attribute__((address_space(3))) const float*  lds_f32_cptr;
typedef __attribute__((address_space(3))) const v16bf*  lds_v16bf_cptr;

// Explicit GLOBAL pointer types so the noinline callee emits global_load/
// global_store (LOADcnt/STOREcnt only) instead of flat_* (LOADcnt+DScnt).
typedef __attribute__((address_space(1))) const __bf16* g_bf16_cptr;
typedef __attribute__((address_space(1))) const v16bf*  g_v16bf_cptr;
typedef __attribute__((address_space(1))) float*        g_f32_ptr;

// ---------------------------------------------------------------------------
// Async global->LDS copy (CDNA5 GLOBAL_LOAD_ASYNC_TO_LDS_B128, ASYNCcnt).
// ---------------------------------------------------------------------------
__device__ __forceinline__ void async_ld16(void* lds, const void* g) {
#if __has_builtin(__builtin_amdgcn_global_load_async_to_lds_b128)
  __builtin_amdgcn_global_load_async_to_lds_b128(
      (gptr128_t)(void*)g, (lptr128_t)lds, /*offset=*/0, /*cpol=*/0);
#else
  *(float4*)lds = *(const float4*)g;   // synchronous fallback
#endif
}

__device__ __forceinline__ void wait_async() {
#if __has_builtin(__builtin_amdgcn_global_load_async_to_lds_b128)
#if __has_builtin(__builtin_amdgcn_s_wait_asynccnt)
  __builtin_amdgcn_s_wait_asynccnt(0);
#else
  asm volatile("s_wait_asynccnt 0" ::: "memory");
#endif
#endif
}

// ---------------------------------------------------------------------------
// Kernel 1: repack conv weights f32 (192,64,1,9) -> bf16 WMMA A-fragments.
// Wbf[((tap*2 + kk)*12 + mtile)*32 + lane][e] ; contiguous K order, same as the B fragments:
// M = lane&15, K = e + ((lane>>4)<<4) (+ kk*32 channel half)
// ---------------------------------------------------------------------------
__global__ void prep_weights_kernel(const float* __restrict__ W,
                                    __bf16* __restrict__ Wbf) {
  int t = blockIdx.x * blockDim.x + threadIdx.x;
  const int total = TAPS * 2 * 12 * 32 * 16;
  if (t >= total) return;
  int e    = t & 15;
  int lane = (t >> 4) & 31;
  int rem  = t >> 9;            // (tap*2 + kk)*12 + mtile
  int m    = rem % 12;
  int tk   = rem / 12;
  int kk   = tk & 1;
  int tap  = tk >> 1;
  int o = m * 16 + (lane & 15);                                     // M row
  int c = e + ((lane >> 4) << 4) + kk * 32; // K = chan
  Wbf[t] = (__bf16)W[(o * C_ + c) * TAPS + tap];
}

// ---------------------------------------------------------------------------
// Half-GEMM: 6 M-tiles (48 acc VGPRs) x 9 taps x 2 K-steps of wmma bf16,
// then gate activations + store. __noinline__ : one code copy, bounded VGPR
// pressure, no cross-call live-range overlap.
// ---------------------------------------------------------------------------
__device__ __noinline__ void gemm_half_and_store(
    lds_bf16_cptr XsT, g_bf16_cptr __restrict__ Wbf, lds_f32_cptr Bias_s,
    g_f32_ptr __restrict__ Zt, g_f32_ptr __restrict__ Ft,
    g_f32_ptr __restrict__ Hout, size_t sb, int tid, int mbase) {
  const int lane  = tid & 31;
  const int wave  = tid >> 5;               // N-tile 0..7
  const int ncol  = (wave << 4) + (lane & 15);
  const int khalf = (lane >> 4) << 4;       // B-frag K half per lane group

  v8f acc[6] = {};
  for (int tap = 0; tap < TAPS; ++tap) {
    lds_bf16_cptr brow = XsT + (ncol + tap) * TSTR;
#pragma unroll
    for (int kk = 0; kk < 2; ++kk) {
      v16bf bfrag = *(lds_v16bf_cptr)(brow + kk * 32 + khalf);
      g_bf16_cptr abase = Wbf + ((size_t)(tap * 2 + kk) * 12 * 32 + lane) * 16
                              + (size_t)mbase * 512;
#pragma unroll
      for (int m = 0; m < 6; ++m) {
        v16bf afrag = *(g_v16bf_cptr)(abase + (size_t)m * 512);
        acc[m] = __builtin_amdgcn_wmma_f32_16x16x32_bf16(
            false, afrag, false, bfrag, (short)0, acc[m], false, false);
      }
    }
  }

  // C/D layout: VGPR r -> M=r (lanes 0-15) / M=r+8 (lanes 16-31).
  const int radd = (lane >> 4) << 3;
#pragma unroll
  for (int m = 0; m < 6; ++m) {
    const int mm   = mbase + m;
    const int gate = mm >> 2;               // wave-uniform: 0=Z, 1=F, 2=O
#pragma unroll
    for (int r = 0; r < 8; ++r) {
      const int o = mm * 16 + r + radd;     // output channel 0..191
      const int h = o & (H_ - 1);
      float y = acc[m][r] + Bias_s[o];
      const size_t idx = (sb * H_ + h) * (size_t)N_ + ncol;
      if (gate == 0)      Zt[idx]   = (y > 0.f) ? y : (__expf(y) - 1.f); // elu
      else if (gate == 1) Ft[idx]   = 1.f / (1.f + __expf(-y));          // sig
      else                Hout[idx] = 1.f / (1.f + __expf(-y));          // sig
    }
  }
}

// ---------------------------------------------------------------------------
// Kernel 2: each workgroup handles samples sb and sb+2048. Sample 1's raw f32
// tile is async-prefetched into LDS while sample 0's GEMM runs.
// ---------------------------------------------------------------------------
__global__ __launch_bounds__(256) void conv_gates_kernel(
    const float* __restrict__ x,       // (S*B, C, N)
    const __bf16* __restrict__ Wbf,
    const float* __restrict__ bias,    // (192)
    float* __restrict__ Zt, float* __restrict__ Ft,
    float* __restrict__ Hout) {        // stash sigmoid(O) in d_out
  __shared__ float Raw[C_ * N_];                                   // 32 KB
  __shared__ __attribute__((aligned(32))) __bf16 XsT[NROWS * TSTR];// 21.25 KB
  __shared__ float Bias_s[H3_];

  const int tid = threadIdx.x;
  const size_t sb0 = blockIdx.x;
  const size_t sb1 = blockIdx.x + (size_t)(S_ * B_ / 2);

  // Kick off async prefetch of sample 1 ASAP (ASYNCcnt-tracked).
  const float* x1 = x + sb1 * (size_t)(C_ * N_);
#pragma unroll
  for (int i = 0; i < 8; ++i) {
    int f4 = (tid + i * 256) * 4;
    async_ld16(&Raw[f4], x1 + f4);
  }

  __builtin_prefetch((const void*)Wbf, 0, 0);   // global_prefetch_b8 (weights)

  if (tid < H3_) Bias_s[tid] = bias[tid];

  // Zero halo rows (conv columns 0..3 and 132..135).
  {
    int idx = tid;
#pragma unroll
    for (int p = 0; p < 2; ++p, idx += 256) {
      int rr  = idx >> 6;                  // 0..7
      int row = (rr < 4) ? rr : (128 + rr);
      XsT[row * TSTR + (idx & 63)] = (__bf16)0.f;
    }
  }

  // Direct-stage sample 0: coalesced global f32 -> bf16 transposed LDS.
  const float* x0 = x + sb0 * (size_t)(C_ * N_);
#pragma unroll
  for (int i = 0; i < 8; ++i) {
    int f4 = tid + i * 256;                // 0..2047 float4 chunks
    int c  = f4 >> 5;
    int n4 = f4 & 31;
    float4 v = *(const float4*)(x0 + c * N_ + n4 * 4);
    XsT[(4 + n4 * 4 + 0) * TSTR + c] = (__bf16)v.x;
    XsT[(4 + n4 * 4 + 1) * TSTR + c] = (__bf16)v.y;
    XsT[(4 + n4 * 4 + 2) * TSTR + c] = (__bf16)v.z;
    XsT[(4 + n4 * 4 + 3) * TSTR + c] = (__bf16)v.w;
  }
  __syncthreads();

  lds_bf16_cptr XsTp  = (lds_bf16_cptr)XsT;
  lds_f32_cptr  BiasP = (lds_f32_cptr)Bias_s;
  g_bf16_cptr   WbfP  = (g_bf16_cptr)Wbf;
  g_f32_ptr     ZtP   = (g_f32_ptr)Zt;
  g_f32_ptr     FtP   = (g_f32_ptr)Ft;
  g_f32_ptr     HoP   = (g_f32_ptr)Hout;

  gemm_half_and_store(XsTp, WbfP, BiasP, ZtP, FtP, HoP, sb0, tid, 0);
  gemm_half_and_store(XsTp, WbfP, BiasP, ZtP, FtP, HoP, sb0, tid, 6);
  __syncthreads();                         // all waves done reading XsT

  // Sample 1: wait only for this wave's own async chunks (self-read),
  // then transpose-convert LDS raw f32 -> bf16 XsT.
  wait_async();
#pragma unroll
  for (int i = 0; i < 8; ++i) {
    int f4 = tid + i * 256;
    int c  = f4 >> 5;
    int n4 = f4 & 31;
    float4 v = *(const float4*)(&Raw[c * N_ + n4 * 4]);
    XsT[(4 + n4 * 4 + 0) * TSTR + c] = (__bf16)v.x;
    XsT[(4 + n4 * 4 + 1) * TSTR + c] = (__bf16)v.y;
    XsT[(4 + n4 * 4 + 2) * TSTR + c] = (__bf16)v.z;
    XsT[(4 + n4 * 4 + 3) * TSTR + c] = (__bf16)v.w;
  }
  __syncthreads();

  gemm_half_and_store(XsTp, WbfP, BiasP, ZtP, FtP, HoP, sb1, tid, 0);
  gemm_half_and_store(XsTp, WbfP, BiasP, ZtP, FtP, HoP, sb1, tid, 6);
}

// ---------------------------------------------------------------------------
// Kernel 3: ForgetMult recurrence over S, float4-vectorized over (b,h,n).
// d_out already holds sigmoid(O); multiply in place, append C_last.
// ---------------------------------------------------------------------------
__global__ __launch_bounds__(256) void recurrence_kernel(
    const float* __restrict__ Zt, const float* __restrict__ Ft,
    const float* __restrict__ hidden, float* __restrict__ out) {
  const size_t i4 = ((size_t)blockIdx.x * blockDim.x + threadIdx.x) * 4;
  const size_t stride = (size_t)B_ * H_ * N_;          // 131072
  float4 c = *(const float4*)(hidden + i4);
  for (int s = 0; s < S_; ++s) {
    const size_t idx = (size_t)s * stride + i4;
    float4 f = *(const float4*)(Ft + idx);
    float4 z = *(const float4*)(Zt + idx);
    float4 o = *(float4*)(out + idx);
    c.x = f.x * z.x + (1.f - f.x) * c.x;  o.x *= c.x;
    c.y = f.y * z.y + (1.f - f.y) * c.y;  o.y *= c.y;
    c.z = f.z * z.z + (1.f - f.z) * c.z;  o.z *= c.z;
    c.w = f.w * z.w + (1.f - f.w) * c.w;  o.w *= c.w;
    *(float4*)(out + idx) = o;            // Hout = sigmoid(O) * C
  }
  *(float4*)(out + (size_t)S_ * stride + i4) = c;      // C_last tail
}

// ---------------------------------------------------------------------------
extern "C" void kernel_launch(void* const* d_in, const int* in_sizes, int n_in,
                              void* d_out, int out_size, void* d_ws, size_t ws_size,
                              hipStream_t stream) {
  const float* x      = (const float*)d_in[0];  // (S,B,C,N)
  const float* hidden = (const float*)d_in[1];  // (B,H,N)
  const float* W      = (const float*)d_in[2];  // (192,64,1,9)
  const float* b      = (const float*)d_in[3];  // (192)
  float* out = (float*)d_out;

  // Workspace layout: [Wbf bf16 fragments | Zt f32 | Ft f32]
  __bf16* Wbf = (__bf16*)d_ws;
  float* Zt = (float*)((char*)d_ws + (size_t)(1u << 18));   // 256KB offset
  float* Ft = Zt + (size_t)S_ * B_ * H_ * N_;

  const int wtotal = TAPS * 2 * 12 * 32 * 16;
  prep_weights_kernel<<<(wtotal + 255) / 256, 256, 0, stream>>>(W, Wbf);
  conv_gates_kernel<<<S_ * B_ / 2, 256, 0, stream>>>(x, Wbf, b, Zt, Ft, out);
  recurrence_kernel<<<(B_ * H_ * N_ / 4 + 255) / 256, 256, 0, stream>>>(
      Zt, Ft, hidden, out);
}
